// GPS_63943473103174
// MI455X (gfx1250) — compile-verified, matrix operand order checked by reading, not run on hardware
//
#include <hip/hip_runtime.h>
#include <hip/hip_bf16.h>
#include <math.h>

#define NN 32768
#define GG 64
#define PERG 512
#define EE 262144
#define HH 128
#define HEADS 4
#define DHD 32

typedef __attribute__((ext_vector_type(16))) _Float16 v16h;
typedef __attribute__((ext_vector_type(8)))  float    v8f;

union AF { v16h v; unsigned u[8]; };
union P2 { unsigned u; _Float16 h[2]; };

__device__ __forceinline__ v8f vzero(){ v8f z = {0.f,0.f,0.f,0.f,0.f,0.f,0.f,0.f}; return z; }

__device__ __forceinline__ int kmap(int r, int g){
  return ((r < 4) ? (2*r) : (16 + 2*(r-4))) + 8*g;
}

// Load a 16x32 f16 A (or 32x16 B from [Nout][K]-transposed weights) fragment.
// Per ISA 7.12.2: lanes 0-15 rows M=0..15 K{0..7,16..23}, lanes 16-31 K{8..15,24..31};
// each VGPR holds a consecutive K-pair -> one dword load each.
__device__ __forceinline__ v16h frag_ld(const _Float16* row, int k0){
  AF f; const int g = (threadIdx.x & 31) >> 4;
#pragma unroll
  for(int r = 0; r < 8; ++r)
    f.u[r] = *reinterpret_cast<const unsigned*>(row + k0 + kmap(r, g));
  return f.v;
}

__device__ __forceinline__ v16h frag_ld_f32(const float* row, int k0){
  AF f; const int g = (threadIdx.x & 31) >> 4;
#pragma unroll
  for(int r = 0; r < 8; ++r){
    const int k = k0 + kmap(r, g);
    float2 d = *reinterpret_cast<const float2*>(row + k);
    P2 p; p.h[0] = (_Float16)d.x; p.h[1] = (_Float16)d.y;
    f.u[r] = p.u;
  }
  return f.v;
}

__device__ __forceinline__ v8f wmma32(v16h a, v16h b, v8f c){
  return __builtin_amdgcn_wmma_f32_16x16x32_f16(false, a, false, b, (short)0, c, false, false);
}

// Stage a 128x32 f16 weight chunk (8KB) into LDS, shared by all waves in the block.
__device__ __forceinline__ void stageB(_Float16* Bs, const _Float16* WT, int K, int c){
  __syncthreads();
  unsigned* bd = reinterpret_cast<unsigned*>(Bs);
  for(int d = threadIdx.x; d < 128*16; d += blockDim.x){
    const int n = d >> 4, kk = d & 15;
    bd[d] = *reinterpret_cast<const unsigned*>(WT + (size_t)n*K + c*32 + kk*2);
  }
  __syncthreads();
}

__device__ __forceinline__ float leaky(float x){ return x > 0.f ? x : 0.01f*x; }

// monotonic float<->uint mapping for atomicMax-based segment_max
__device__ __forceinline__ unsigned orderf(float x){
  unsigned u = __float_as_uint(x);
  return (u >> 31) ? ~u : (u | 0x80000000u);
}
__device__ __forceinline__ float unorderf(unsigned u){
  unsigned b = (u >> 31) ? (u & 0x7fffffffu) : ~u;
  return __uint_as_float(b);
}

// ----------------------------- prep kernels -----------------------------

__global__ void k_init(unsigned* aggU, float* stats){
  const int i = blockIdx.x*256 + threadIdx.x;
  if(i < NN*HH) aggU[i] = 0x007FFFFFu;          // orderf(-inf)
  if(i < 12*HH) stats[i] = 0.f;
}

__global__ void k_convT(const float* W, _Float16* WT, int K, int Nout){
  const int i = blockIdx.x*256 + threadIdx.x;
  if(i >= K*Nout) return;
  const int n = i / K, k = i % K;
  WT[i] = (_Float16)W[(size_t)k*Nout + n];
}

__global__ void k_prep_z(const float* nf, const float* hid, _Float16* z16, _Float16* n16){
  const int i = blockIdx.x*256 + threadIdx.x;
  if(i >= NN*HH) return;
  const int n = i >> 7, c = i & 127;
  const float a = nf[i];
  n16[i] = (_Float16)a;
  z16[(size_t)n*256 + c]       = (_Float16)a;
  z16[(size_t)n*256 + 128 + c] = (_Float16)hid[i];
}

__global__ void k_prep_g(const float* gf, _Float16* g16){
  const int i = blockIdx.x*256 + threadIdx.x;
  if(i < GG*HH) g16[i] = (_Float16)gf[i];
}

// ----------------------- fused edge MLP + segment_max -----------------------
// 64 edges/block, 4 waves (16 edges each). ea GEMM -> LDS, 4-layer MLP via WMMA,
// leaky between layers via LDS C->A relayout, atomicMax scatter at the end.
__global__ __launch_bounds__(128) void k_edge(
    const float* __restrict__ ea_in, const int* __restrict__ srcI, const int* __restrict__ dstI,
    const float* __restrict__ b_ee, const _Float16* __restrict__ z16, const _Float16* __restrict__ g16,
    const _Float16* Weet, const _Float16* M0t, const _Float16* M1t,
    const _Float16* M2t, const _Float16* M3t, unsigned* aggU)
{
  __shared__ _Float16 Bs[128*32];     // 8KB weight chunk
  __shared__ _Float16 eaS[64*HH];     // 16KB ea tile (f16)
  __shared__ _Float16 actS[64*HH];    // 16KB activation tile (f16)
  const int tid = threadIdx.x, lane = tid & 31, w = tid >> 5;
  const int ln = lane & 15, lg = lane >> 4;
  const int ebase = blockIdx.x * 64;
  const int rw = w * 16;
  const int eA = ebase + rw + ln;
  const int myDst = dstI[eA];
  const int mySrc = srcI[eA];
  const int myG = myDst >> 9;         // batch[dst] = dst / 512

  v8f acc[8];
#pragma unroll
  for(int n = 0; n < 8; ++n) acc[n] = vzero();

  // ---- ea = edge_attr @ W_ee + b_ee  (K=128) ----
  for(int c = 0; c < 4; ++c){
    stageB(Bs, Weet, 128, c);
    v16h a = frag_ld_f32(ea_in + (size_t)eA*HH, c*32);
#pragma unroll
    for(int n = 0; n < 8; ++n)
      acc[n] = wmma32(a, frag_ld(Bs + (n*16 + ln)*32, 0), acc[n]);
  }
#pragma unroll
  for(int nt = 0; nt < 8; ++nt){
    const int col = nt*16 + ln;
    const float bb = b_ee[col];
#pragma unroll
    for(int r = 0; r < 8; ++r)
      eaS[(rw + r + 8*lg)*HH + col] = (_Float16)(acc[nt][r] + bb);
  }

  // ---- layer 0: cat(z[dst], z[src], ea, gfi) @ M0   (K=768) ----
#pragma unroll
  for(int n = 0; n < 8; ++n) acc[n] = vzero();
  for(int c = 0; c < 24; ++c){
    stageB(Bs, M0t, 768, c);
    const int k0 = c*32;
    v16h a;
    if(k0 < 256)      a = frag_ld(z16 + (size_t)myDst*256, k0);
    else if(k0 < 512) a = frag_ld(z16 + (size_t)mySrc*256, k0 - 256);
    else if(k0 < 640) a = frag_ld(eaS + (rw + ln)*HH,      k0 - 512);
    else              a = frag_ld(g16 + (size_t)myG*HH,    k0 - 640);
#pragma unroll
    for(int n = 0; n < 8; ++n)
      acc[n] = wmma32(a, frag_ld(Bs + (n*16 + ln)*32, 0), acc[n]);
  }
#pragma unroll
  for(int nt = 0; nt < 8; ++nt)
#pragma unroll
    for(int r = 0; r < 8; ++r)
      actS[(rw + r + 8*lg)*HH + nt*16 + ln] = (_Float16)leaky(acc[nt][r]);

  // ---- layers 1..3 (K=128 each), last layer scatters with atomicMax ----
  for(int L = 0; L < 3; ++L){
    const _Float16* Wt = (L == 0) ? M1t : ((L == 1) ? M2t : M3t);
#pragma unroll
    for(int n = 0; n < 8; ++n) acc[n] = vzero();
    for(int c = 0; c < 4; ++c){
      stageB(Bs, Wt, 128, c);
      v16h a = frag_ld(actS + (rw + ln)*HH, c*32);
#pragma unroll
      for(int n = 0; n < 8; ++n)
        acc[n] = wmma32(a, frag_ld(Bs + (n*16 + ln)*32, 0), acc[n]);
    }
    if(L < 2){
#pragma unroll
      for(int nt = 0; nt < 8; ++nt)
#pragma unroll
        for(int r = 0; r < 8; ++r)
          actS[(rw + r + 8*lg)*HH + nt*16 + ln] = (_Float16)leaky(acc[nt][r]);
    } else {
      int erow[8];
#pragma unroll
      for(int r = 0; r < 8; ++r) erow[r] = dstI[ebase + rw + r + 8*lg];
#pragma unroll
      for(int nt = 0; nt < 8; ++nt){
        const int col = nt*16 + ln;
#pragma unroll
        for(int r = 0; r < 8; ++r)
          atomicMax(&aggU[(size_t)erow[r]*HH + col], orderf(acc[nt][r]));
      }
    }
  }
}

__global__ void k_decode(const unsigned* aggU, _Float16* agg16){
  const int i = blockIdx.x*256 + threadIdx.x;
  if(i >= NN*HH) return;
  const unsigned u = aggU[i];
  const float v = (u == 0x007FFFFFu) ? 0.f : unorderf(u);  // empty segment -> 0
  agg16[i] = (_Float16)v;
}

// ---------------- h_local = z@U1 + agg@U2 ; t1 = +node_fts ; bn1 stats ----------------
__global__ __launch_bounds__(256) void k_hlocal(
    const _Float16* z16, const _Float16* agg16, const _Float16* U1t, const _Float16* U2t,
    const float* node_fts, float* t1, float* stats)
{
  __shared__ _Float16 Bs[128*32];
  const int tid = threadIdx.x, lane = tid & 31, w = tid >> 5;
  const int ln = lane & 15, lg = lane >> 4;
  const int rbase = blockIdx.x*128 + w*16;
  v8f acc[8];
#pragma unroll
  for(int n = 0; n < 8; ++n) acc[n] = vzero();
  for(int c = 0; c < 8; ++c){
    stageB(Bs, U1t, 256, c);
    v16h a = frag_ld(z16 + (size_t)(rbase + ln)*256, c*32);
#pragma unroll
    for(int n = 0; n < 8; ++n)
      acc[n] = wmma32(a, frag_ld(Bs + (n*16 + ln)*32, 0), acc[n]);
  }
  for(int c = 0; c < 4; ++c){
    stageB(Bs, U2t, 128, c);
    v16h a = frag_ld(agg16 + (size_t)(rbase + ln)*HH, c*32);
#pragma unroll
    for(int n = 0; n < 8; ++n)
      acc[n] = wmma32(a, frag_ld(Bs + (n*16 + ln)*32, 0), acc[n]);
  }
#pragma unroll
  for(int nt = 0; nt < 8; ++nt){
    const int col = nt*16 + ln;
    float s = 0.f, q = 0.f;
#pragma unroll
    for(int r = 0; r < 8; ++r){
      const int row = rbase + r + 8*lg;
      const float v = acc[nt][r] + node_fts[(size_t)row*HH + col];
      t1[(size_t)row*HH + col] = v;
      s += v; q += v*v;
    }
    s += __shfl_xor(s, 16, 32); q += __shfl_xor(q, 16, 32);
    if(lane < 16){ atomicAdd(&stats[0*128 + col], s); atomicAdd(&stats[1*128 + col], q); }
  }
}

// ---------------- qkv projection + head-layout scatter ----------------
__global__ __launch_bounds__(256) void k_qkv(
    const _Float16* node16, const _Float16* Wqkvt, const float* bqkv,
    _Float16* qb, _Float16* kb, _Float16* vT)
{
  __shared__ _Float16 Bs[128*32];
  const int tid = threadIdx.x, lane = tid & 31, w = tid >> 5;
  const int ln = lane & 15, lg = lane >> 4;
  const int rbase = blockIdx.x*128 + w*16;
  for(int grp = 0; grp < 3; ++grp){
    v8f acc[8];
#pragma unroll
    for(int n = 0; n < 8; ++n) acc[n] = vzero();
    for(int c = 0; c < 4; ++c){
      stageB(Bs, Wqkvt + (size_t)grp*128*128, 128, c);
      v16h a = frag_ld(node16 + (size_t)(rbase + ln)*HH, c*32);
#pragma unroll
      for(int n = 0; n < 8; ++n)
        acc[n] = wmma32(a, frag_ld(Bs + (n*16 + ln)*32, 0), acc[n]);
    }
#pragma unroll
    for(int nt = 0; nt < 8; ++nt){
      const int col = nt*16 + ln;
      const float bb = bqkv[grp*128 + col];
#pragma unroll
      for(int r = 0; r < 8; ++r){
        const int node = rbase + r + 8*lg;
        const int head = col >> 5, dh = col & 31;
        const int g = node >> 9, p = node & 511, gh = g*4 + head;
        const _Float16 hv = (_Float16)(acc[nt][r] + bb);
        if(grp == 0)      qb[((size_t)gh*PERG + p)*DHD + dh] = hv;
        else if(grp == 1) kb[((size_t)gh*PERG + p)*DHD + dh] = hv;
        else              vT[((size_t)gh*DHD + dh)*PERG + p] = hv;
      }
    }
  }
}

// ---------------- attention: one wave per (graph,head,16-query tile) ----------------
__global__ __launch_bounds__(32) void k_attn(
    const _Float16* qb, const _Float16* kb, const _Float16* vT, _Float16* of16)
{
  __shared__ float    scS[16*PERG];   // 32KB scores
  __shared__ _Float16 prS[16*PERG];   // 16KB probs
  const int lane = threadIdx.x & 31, ln = lane & 15, lg = lane >> 4;
  const int qt = blockIdx.x & 31;
  const int gh = blockIdx.x >> 5;
  const int g = gh >> 2, h = gh & 3;
  const float scale = 0.17677669529663687f;   // 1/sqrt(32)

  v16h aq = frag_ld(qb + ((size_t)gh*PERG + qt*16 + ln)*DHD, 0);
  for(int kt = 0; kt < 32; ++kt){
    v16h bk = frag_ld(kb + ((size_t)gh*PERG + kt*16 + ln)*DHD, 0);
    v8f s = wmma32(aq, bk, vzero());
#pragma unroll
    for(int r = 0; r < 8; ++r)
      scS[(r + 8*lg)*PERG + kt*16 + ln] = s[r]*scale;
  }
  for(int m = 0; m < 16; ++m){
    float mx = -1e30f;
    for(int j = lane; j < PERG; j += 32) mx = fmaxf(mx, scS[m*PERG + j]);
#pragma unroll
    for(int o = 16; o > 0; o >>= 1) mx = fmaxf(mx, __shfl_xor(mx, o, 32));
    float sm = 0.f;
    for(int j = lane; j < PERG; j += 32){
      const float e = __expf(scS[m*PERG + j] - mx);
      scS[m*PERG + j] = e; sm += e;
    }
#pragma unroll
    for(int o = 16; o > 0; o >>= 1) sm += __shfl_xor(sm, o, 32);
    const float inv = 1.f/sm;
    for(int j = lane; j < PERG; j += 32)
      prS[m*PERG + j] = (_Float16)(scS[m*PERG + j]*inv);
  }
  v8f o0 = vzero(), o1 = vzero();
  for(int kc = 0; kc < 16; ++kc){
    v16h a = frag_ld(prS + ln*PERG, kc*32);
    o0 = wmma32(a, frag_ld(vT + ((size_t)gh*DHD + ln)*PERG,      kc*32), o0);
    o1 = wmma32(a, frag_ld(vT + ((size_t)gh*DHD + 16 + ln)*PERG, kc*32), o1);
  }
#pragma unroll
  for(int r = 0; r < 8; ++r){
    const int node = g*PERG + qt*16 + r + 8*lg;
    of16[(size_t)node*HH + h*32 + ln]      = (_Float16)o0[r];
    of16[(size_t)node*HH + h*32 + 16 + ln] = (_Float16)o1[r];
  }
}

// ---------------- t2 = o@Wo + bo + node_fts ; bn2 stats ----------------
__global__ __launch_bounds__(256) void k_proj(
    const _Float16* of16, const _Float16* Wot, const float* bo,
    const float* node_fts, float* t2, float* stats)
{
  __shared__ _Float16 Bs[128*32];
  const int tid = threadIdx.x, lane = tid & 31, w = tid >> 5;
  const int ln = lane & 15, lg = lane >> 4;
  const int rbase = blockIdx.x*128 + w*16;
  v8f acc[8];
#pragma unroll
  for(int n = 0; n < 8; ++n) acc[n] = vzero();
  for(int c = 0; c < 4; ++c){
    stageB(Bs, Wot, 128, c);
    v16h a = frag_ld(of16 + (size_t)(rbase + ln)*HH, c*32);
#pragma unroll
    for(int n = 0; n < 8; ++n)
      acc[n] = wmma32(a, frag_ld(Bs + (n*16 + ln)*32, 0), acc[n]);
  }
#pragma unroll
  for(int nt = 0; nt < 8; ++nt){
    const int col = nt*16 + ln;
    const float bb = bo[col];
    float s = 0.f, q = 0.f;
#pragma unroll
    for(int r = 0; r < 8; ++r){
      const int row = rbase + r + 8*lg;
      const float v = acc[nt][r] + bb + node_fts[(size_t)row*HH + col];
      t2[(size_t)row*HH + col] = v;
      s += v; q += v*v;
    }
    s += __shfl_xor(s, 16, 32); q += __shfl_xor(q, 16, 32);
    if(lane < 16){ atomicAdd(&stats[2*128 + col], s); atomicAdd(&stats[3*128 + col], q); }
  }
}

__global__ void k_bnfin(const float* w, const float* b, float* stats, int idxSum, int idxSc){
  const int c = threadIdx.x;
  if(c >= 128) return;
  const float mean = stats[idxSum*128 + c] / (float)NN;
  const float var  = stats[(idxSum+1)*128 + c] / (float)NN - mean*mean;
  const float sc   = w[c]*rsqrtf(var + 1e-5f);
  stats[idxSc*128 + c]     = sc;
  stats[(idxSc+1)*128 + c] = b[c] - mean*sc;
}

__global__ void k_combine(const float* t1, const float* t2, const float* stats,
                          float* outf, _Float16* out16){
  const int i = blockIdx.x*256 + threadIdx.x;
  if(i >= NN*HH) return;
  const int c = i & 127;
  const float v = t1[i]*stats[6*128 + c] + stats[7*128 + c]
                + t2[i]*stats[8*128 + c] + stats[9*128 + c];
  outf[i] = v;
  out16[i] = (_Float16)v;
}

// ---------------- r = relu(out@W1 + b1) ----------------
__global__ __launch_bounds__(256) void k_mlp1(
    const _Float16* out16, const _Float16* W1t, const float* b1, _Float16* r16)
{
  __shared__ _Float16 Bs[128*32];
  const int tid = threadIdx.x, lane = tid & 31, w = tid >> 5;
  const int ln = lane & 15, lg = lane >> 4;
  const int rbase = blockIdx.x*128 + w*16;
  for(int half = 0; half < 2; ++half){
    v8f acc[8];
#pragma unroll
    for(int n = 0; n < 8; ++n) acc[n] = vzero();
    for(int c = 0; c < 4; ++c){
      stageB(Bs, W1t + (size_t)half*128*128, 128, c);
      v16h a = frag_ld(out16 + (size_t)(rbase + ln)*HH, c*32);
#pragma unroll
      for(int n = 0; n < 8; ++n)
        acc[n] = wmma32(a, frag_ld(Bs + (n*16 + ln)*32, 0), acc[n]);
    }
#pragma unroll
    for(int nt = 0; nt < 8; ++nt){
      const int col = nt*16 + ln;
      const float bb = b1[half*128 + col];
#pragma unroll
      for(int r = 0; r < 8; ++r){
        const int row = rbase + r + 8*lg;
        r16[(size_t)row*256 + half*128 + col] = (_Float16)fmaxf(acc[nt][r] + bb, 0.f);
      }
    }
  }
}

// ---------------- y = r@W2 + b2 + out ; bn3 stats ----------------
__global__ __launch_bounds__(256) void k_mlp2(
    const _Float16* r16, const _Float16* W2t, const float* b2,
    const float* outf, float* y, float* stats)
{
  __shared__ _Float16 Bs[128*32];
  const int tid = threadIdx.x, lane = tid & 31, w = tid >> 5;
  const int ln = lane & 15, lg = lane >> 4;
  const int rbase = blockIdx.x*128 + w*16;
  v8f acc[8];
#pragma unroll
  for(int n = 0; n < 8; ++n) acc[n] = vzero();
  for(int c = 0; c < 8; ++c){
    stageB(Bs, W2t, 256, c);
    v16h a = frag_ld(r16 + (size_t)(rbase + ln)*256, c*32);
#pragma unroll
    for(int n = 0; n < 8; ++n)
      acc[n] = wmma32(a, frag_ld(Bs + (n*16 + ln)*32, 0), acc[n]);
  }
#pragma unroll
  for(int nt = 0; nt < 8; ++nt){
    const int col = nt*16 + ln;
    const float bb = b2[col];
    float s = 0.f, q = 0.f;
#pragma unroll
    for(int r = 0; r < 8; ++r){
      const int row = rbase + r + 8*lg;
      const float v = acc[nt][r] + bb + outf[(size_t)row*HH + col];
      y[(size_t)row*HH + col] = v;
      s += v; q += v*v;
    }
    s += __shfl_xor(s, 16, 32); q += __shfl_xor(q, 16, 32);
    if(lane < 16){ atomicAdd(&stats[4*128 + col], s); atomicAdd(&stats[5*128 + col], q); }
  }
}

__global__ void k_final(const float* y, const float* stats, float* out){
  const int i = blockIdx.x*256 + threadIdx.x;
  if(i >= NN*HH) return;
  const int c = i & 127;
  out[i] = y[i]*stats[10*128 + c] + stats[11*128 + c];
}

// ----------------------------- host launch -----------------------------

extern "C" void kernel_launch(void* const* d_in, const int* in_sizes, int n_in,
                              void* d_out, int out_size, void* d_ws, size_t ws_size,
                              hipStream_t stream) {
  const float* node_fts  = (const float*)d_in[0];
  const float* edge_attr = (const float*)d_in[1];
  const float* graph_fts = (const float*)d_in[2];
  const float* hidden    = (const float*)d_in[3];
  const float* W_ee = (const float*)d_in[4];
  const float* b_ee = (const float*)d_in[5];
  const float* M0   = (const float*)d_in[6];
  const float* M1   = (const float*)d_in[7];
  const float* M2   = (const float*)d_in[8];
  const float* M3   = (const float*)d_in[9];
  const float* U1   = (const float*)d_in[10];
  const float* U2   = (const float*)d_in[11];
  const float* Wqkv = (const float*)d_in[12];
  const float* bqkv = (const float*)d_in[13];
  const float* Wo   = (const float*)d_in[14];
  const float* bo   = (const float*)d_in[15];
  const float* bn1w = (const float*)d_in[16];
  const float* bn1b = (const float*)d_in[17];
  const float* bn2w = (const float*)d_in[18];
  const float* bn2b = (const float*)d_in[19];
  const float* bn3w = (const float*)d_in[20];
  const float* bn3b = (const float*)d_in[21];
  const float* W1   = (const float*)d_in[22];
  const float* b1   = (const float*)d_in[23];
  const float* W2   = (const float*)d_in[24];
  const float* b2   = (const float*)d_in[25];
  const int* ei     = (const int*)d_in[26];
  const int* srcI = ei;
  const int* dstI = ei + EE;

  char* ws = (char*)d_ws;
  size_t off = 0;
  auto alloc = [&](size_t bytes) -> char* {
    off = (off + 255) & ~(size_t)255;
    char* p = ws + off;
    off += bytes;
    return p;
  };

  _Float16* z16    = (_Float16*)alloc((size_t)NN*256*2);
  _Float16* node16 = (_Float16*)alloc((size_t)NN*HH*2);
  _Float16* g16    = (_Float16*)alloc((size_t)GG*HH*2);
  _Float16* Weet   = (_Float16*)alloc(128*128*2);
  _Float16* M0t    = (_Float16*)alloc(768*128*2);
  _Float16* M1t    = (_Float16*)alloc(128*128*2);
  _Float16* M2t    = (_Float16*)alloc(128*128*2);
  _Float16* M3t    = (_Float16*)alloc(128*128*2);
  _Float16* U1t    = (_Float16*)alloc(256*128*2);
  _Float16* U2t    = (_Float16*)alloc(128*128*2);
  _Float16* Wqkvt  = (_Float16*)alloc(384*128*2);
  _Float16* Wot    = (_Float16*)alloc(128*128*2);
  _Float16* W1t    = (_Float16*)alloc(256*128*2);
  _Float16* W2t    = (_Float16*)alloc(256*128*2);
  unsigned* aggU   = (unsigned*)alloc((size_t)NN*HH*4);
  _Float16* agg16  = (_Float16*)alloc((size_t)NN*HH*2);
  float*    t1     = (float*)   alloc((size_t)NN*HH*4);
  float*    t2     = (float*)   alloc((size_t)NN*HH*4);
  _Float16* qb     = (_Float16*)alloc((size_t)NN*HH*2);
  _Float16* kb     = (_Float16*)alloc((size_t)NN*HH*2);
  _Float16* vT     = (_Float16*)alloc((size_t)NN*HH*2);
  _Float16* of16   = (_Float16*)alloc((size_t)NN*HH*2);
  float*    outf   = (float*)   alloc((size_t)NN*HH*4);
  _Float16* out16  = (_Float16*)alloc((size_t)NN*HH*2);
  _Float16* r16    = (_Float16*)alloc((size_t)NN*256*2);
  float*    stats  = (float*)   alloc(12*128*4);
  float*    y      = t1;        // t1 dead after k_combine

  const int NH = NN*HH;
  k_init  <<<(NH + 255)/256, 256, 0, stream>>>(aggU, stats);
  k_prep_z<<<(NH + 255)/256, 256, 0, stream>>>(node_fts, hidden, z16, node16);
  k_prep_g<<<(GG*HH + 255)/256, 256, 0, stream>>>(graph_fts, g16);

  auto convT = [&](const float* W, _Float16* WT, int K, int Nout){
    const int tot = K*Nout;
    k_convT<<<(tot + 255)/256, 256, 0, stream>>>(W, WT, K, Nout);
  };
  convT(W_ee, Weet, 128, 128);
  convT(M0,   M0t,  768, 128);
  convT(M1,   M1t,  128, 128);
  convT(M2,   M2t,  128, 128);
  convT(M3,   M3t,  128, 128);
  convT(U1,   U1t,  256, 128);
  convT(U2,   U2t,  128, 128);
  convT(Wqkv, Wqkvt,128, 384);
  convT(Wo,   Wot,  128, 128);
  convT(W1,   W1t,  128, 256);
  convT(W2,   W2t,  256, 128);

  k_edge  <<<EE/64, 128, 0, stream>>>(edge_attr, srcI, dstI, b_ee, z16, g16,
                                      Weet, M0t, M1t, M2t, M3t, aggU);
  k_decode<<<(NH + 255)/256, 256, 0, stream>>>(aggU, agg16);
  k_hlocal<<<NN/128, 256, 0, stream>>>(z16, agg16, U1t, U2t, node_fts, t1, stats);
  k_qkv   <<<NN/128, 256, 0, stream>>>(node16, Wqkvt, bqkv, qb, kb, vT);
  k_attn  <<<GG*HEADS*(PERG/16), 32, 0, stream>>>(qb, kb, vT, of16);
  k_proj  <<<NN/128, 256, 0, stream>>>(of16, Wot, bo, node_fts, t2, stats);
  k_bnfin <<<1, 128, 0, stream>>>(bn1w, bn1b, stats, 0, 6);
  k_bnfin <<<1, 128, 0, stream>>>(bn2w, bn2b, stats, 2, 8);
  k_combine<<<(NH + 255)/256, 256, 0, stream>>>(t1, t2, stats, outf, out16);
  k_mlp1  <<<NN/128, 256, 0, stream>>>(out16, W1t, b1, r16);
  k_mlp2  <<<NN/128, 256, 0, stream>>>(r16, W2t, b2, outf, y, stats);
  k_bnfin <<<1, 128, 0, stream>>>(bn3w, bn3b, stats, 4, 10);
  k_final <<<(NH + 255)/256, 256, 0, stream>>>(y, stats, (float*)d_out);
}
